// RNN_39067022524663
// MI455X (gfx1250) — matrix operand order repaired and checked by reading, not compiled, not run on hardware
//
#include <hip/hip_runtime.h>
#include <hip/hip_bf16.h>

typedef __attribute__((ext_vector_type(16))) __bf16 v16bf;
typedef __attribute__((ext_vector_type(8)))  __bf16 v8bf;
typedef __attribute__((ext_vector_type(8)))  float  v8f;
typedef int v4i __attribute__((vector_size(16)));

#define B_ROWS  32768
#define D_IN    512
#define D_DENSE 256
#define T_STEPS 16
#define H1      128
#define H2      32

// ---- workspace fragment layout (bf16 elements), fragment-linear:
// frag f = kchunk*ntiles + ntile ; within frag: lane*16 + e
// B-frag element (lane L, e): N = ntile*16 + (L&15), K = kchunk*32 + (e>>3)*16 + (L>>4)*8 + (e&7)
#define OFF_WD1 0
#define SZ_WD1  (256*512)            // 16 kchunks x 16 ntiles
#define OFF_WX1 (OFF_WD1 + SZ_WD1)
#define SZ_WX1  (32*512)             // 1 kchunk (K=16 padded to 32) x 32 ntiles
#define OFF_WH1 (OFF_WX1 + SZ_WX1)
#define SZ_WH1  (128*512)            // 4 kchunks x 32 ntiles
#define OFF_WX2 (OFF_WH1 + SZ_WH1)
#define SZ_WX2  (32*512)             // 4 kchunks x 8 ntiles
#define OFF_WH2 (OFF_WX2 + SZ_WX2)
#define SZ_WH2  (8*512)              // 1 kchunk x 8 ntiles
#define TOTAL_FRAG_ELEMS (OFF_WH2 + SZ_WH2)

// ---- LDS layout (bf16 elements)
#define LW_WX1 0
#define LW_WH1 (LW_WX1 + SZ_WX1)
#define LW_WX2 (LW_WH1 + SZ_WH1)
#define LW_WH2 (LW_WX2 + SZ_WX2)
#define WEIGHT_ELEMS (SZ_WX1 + SZ_WH1 + SZ_WX2 + SZ_WH2)   // 102400 (200 KB)

#define DSO_S 272                   // padded stride for dense-out [16][256]
#define H1_S  136                   // padded stride for h1 [16][128]
#define H2_S  40                    // padded stride for h2 [16][32]
#define WAVE_ELEMS (16*DSO_S + 16*H1_S + 16*H2_S)           // 7168
#define NWAVES 8
#define SMEM_BYTES ((WEIGHT_ELEMS + NWAVES*WAVE_ELEMS) * 2) // 319488 B < 320 KB WGP LDS

// CDNA5 async global->LDS path (ASYNCcnt), guarded so compilation can't regress
#if defined(__gfx1250__) && __has_builtin(__builtin_amdgcn_global_load_async_to_lds_b128) && __has_builtin(__builtin_amdgcn_s_wait_asynccnt)
#define HAVE_ASYNC_LDS 1
#else
#define HAVE_ASYNC_LDS 0
#endif

// ---------------- prep: fp32 weights -> bf16 fragment-linear ----------------
__device__ __forceinline__ void frag_fill(__bf16* dst, const float* W, int Krows,
                                          int Ncols, int ntiles, int idx) {
  int f = idx >> 9;
  int s = idx & 511;
  int lane = s >> 4, e = s & 15;
  int kc = f / ntiles, nt = f - kc * ntiles;
  int n = nt * 16 + (lane & 15);
  int k = kc * 32 + (e >> 3) * 16 + (lane >> 4) * 8 + (e & 7);
  float v = (k < Krows) ? W[(size_t)k * Ncols + n] : 0.0f;
  dst[idx] = (__bf16)v;
}

__global__ void prep_weights(const float* __restrict__ Wd1, const float* __restrict__ Wx1,
                             const float* __restrict__ Wh1, const float* __restrict__ Wx2,
                             const float* __restrict__ Wh2, __bf16* __restrict__ wf) {
  int idx = blockIdx.x * blockDim.x + threadIdx.x;
  if (idx < SZ_WD1) { frag_fill(wf + OFF_WD1, Wd1, 512, 256, 16, idx); return; }
  idx -= SZ_WD1;
  if (idx < SZ_WX1) { frag_fill(wf + OFF_WX1, Wx1, 16, 512, 32, idx); return; }
  idx -= SZ_WX1;
  if (idx < SZ_WH1) { frag_fill(wf + OFF_WH1, Wh1, 128, 512, 32, idx); return; }
  idx -= SZ_WH1;
  if (idx < SZ_WX2) { frag_fill(wf + OFF_WX2, Wx2, 128, 128, 8, idx); return; }
  idx -= SZ_WX2;
  if (idx < SZ_WH2) { frag_fill(wf + OFF_WH2, Wh2, 32, 128, 8, idx); }
}

// ---------------- fragment loaders ----------------
__device__ __forceinline__ v8f wmma_bf16(v16bf a, v16bf b, v8f c) {
  return __builtin_amdgcn_wmma_f32_16x16x32_bf16(false, a, false, b, (short)0, c,
                                                 false, false);
}

// B-fragment from fragment-linear storage (global)
__device__ __forceinline__ v16bf gB(const __bf16* base, int f, int lane) {
  const v8bf* p = (const v8bf*)(base + (size_t)f * 512 + lane * 16);
  v8bf lo = p[0], hi = p[1];
  v16bf r;
#pragma unroll
  for (int e = 0; e < 8; ++e) { r[e] = lo[e]; r[8 + e] = hi[e]; }
  return r;
}
// B-fragment from fragment-linear storage (LDS)
__device__ __forceinline__ v16bf ldsB(const __bf16* base, int f, int lane) {
  const v8bf* p = (const v8bf*)(base + f * 512 + lane * 16);
  v8bf lo = p[0], hi = p[1];
  v16bf r;
#pragma unroll
  for (int e = 0; e < 8; ++e) { r[e] = lo[e]; r[8 + e] = hi[e]; }
  return r;
}

// A-fragment (16x32) from row-major fp32 global: K runs contiguous per ISA layout
__device__ __forceinline__ v16bf gA_f32(const float* base, int ld, int k0, int lane) {
  int row = lane & 15, half = lane >> 4;
  const float* p = base + (size_t)row * ld + k0 + half * 8;
  float4 a0 = *(const float4*)(p);
  float4 a1 = *(const float4*)(p + 4);
  float4 b0 = *(const float4*)(p + 16);
  float4 b1 = *(const float4*)(p + 20);
  v16bf r;
  r[0]=(__bf16)a0.x; r[1]=(__bf16)a0.y; r[2]=(__bf16)a0.z; r[3]=(__bf16)a0.w;
  r[4]=(__bf16)a1.x; r[5]=(__bf16)a1.y; r[6]=(__bf16)a1.z; r[7]=(__bf16)a1.w;
  r[8]=(__bf16)b0.x; r[9]=(__bf16)b0.y; r[10]=(__bf16)b0.z; r[11]=(__bf16)b0.w;
  r[12]=(__bf16)b1.x; r[13]=(__bf16)b1.y; r[14]=(__bf16)b1.z; r[15]=(__bf16)b1.w;
  return r;
}

// A-fragment (16x32) from row-major bf16 LDS; `full`=false zero-pads K>=16
__device__ __forceinline__ v16bf ldsA(const __bf16* buf, int stride, int col0,
                                      bool full, int lane) {
  int row = lane & 15, half = lane >> 4;
  const __bf16* p = buf + row * stride + col0 + half * 8;
  v8bf lo = *(const v8bf*)p;
  v16bf r;
#pragma unroll
  for (int e = 0; e < 8; ++e) r[e] = lo[e];
  if (full) {
    v8bf hi = *(const v8bf*)(p + 16);
#pragma unroll
    for (int e = 0; e < 8; ++e) r[8 + e] = hi[e];
  } else {
#pragma unroll
    for (int e = 0; e < 8; ++e) r[8 + e] = (__bf16)0.0f;
  }
  return r;
}

__device__ __forceinline__ float sigm(float x) { return 1.0f / (1.0f + __expf(-x)); }

// Fast tanh: prefer CDNA5 V_TANH transcendental, else 1-transcendental rational.
__device__ __forceinline__ float fast_tanh(float x) {
#if defined(__gfx1250__) && __has_builtin(__builtin_amdgcn_tanhf)
  return __builtin_amdgcn_tanhf(x);
#else
  float e = __expf(2.0f * x);
  return (e - 1.0f) / (e + 1.0f);
#endif
}

// ---------------- fused forward ----------------
__global__ __launch_bounds__(256) void rnn_fused(
    const float* __restrict__ x, const float* __restrict__ b_d1,
    const float* __restrict__ bn_gamma, const float* __restrict__ bn_beta,
    const float* __restrict__ bn_mean, const float* __restrict__ bn_var,
    const float* __restrict__ b1, const float* __restrict__ b2,
    const float* __restrict__ W_d2, const float* __restrict__ b_d2,
    const __bf16* __restrict__ wf, float* __restrict__ out) {
  extern __shared__ __align__(16) __bf16 sm[];
  const int tid = threadIdx.x;
  const int lane = tid & 31;
  const int wave = tid >> 5;
  const int half = lane >> 4;
  const int ln = lane & 15;

  // ---- stage LSTM weight fragments into LDS (shared by all 8 waves) ----
  // Async DMA (ASYNCcnt) so it overlaps the Dense1 GEMM below, which only
  // touches global memory and this wave's private LDS scratch.
#if HAVE_ASYNC_LDS
  {
    v4i* gsrc = (v4i*)(wf + OFF_WX1);
    __attribute__((address_space(3))) v4i* ldst =
        (__attribute__((address_space(3))) v4i*)sm;
    for (int i = tid; i < (WEIGHT_ELEMS * 2) / 16; i += blockDim.x)
      __builtin_amdgcn_global_load_async_to_lds_b128(gsrc + i, ldst + i, 0, 0);
  }
#else
  {
    const uint4* src = (const uint4*)(wf + OFF_WX1);
    uint4* dst = (uint4*)sm;
    for (int i = tid; i < WEIGHT_ELEMS / 8; i += blockDim.x) dst[i] = src[i];
  }
#endif

  __bf16* dso = sm + WEIGHT_ELEMS + wave * WAVE_ELEMS;  // dense out [16][256] pad 272
  __bf16* h1b = dso + 16 * DSO_S;                       // h1 [16][128] pad 136
  __bf16* h2b = h1b + 16 * H1_S;                        // h2 [16][32]  pad 40
  for (int i = lane; i < 16 * H1_S + 16 * H2_S; i += 32) h1b[i] = (__bf16)0.0f;

  const int r0 = blockIdx.x * (NWAVES * 16) + wave * 16;
  const float* xrow = x + (size_t)r0 * D_IN;

  // ---- Dense(512->256) + BatchNorm + LeakyReLU(0.2): overlaps weight DMA ----
  v16bf xf[16];
#pragma unroll
  for (int kc = 0; kc < 16; ++kc) xf[kc] = gA_f32(xrow, D_IN, kc * 32, lane);

  for (int nt = 0; nt < 16; ++nt) {
    v8f acc;
#pragma unroll
    for (int e = 0; e < 8; ++e) acc[e] = 0.0f;
#pragma unroll
    for (int kc = 0; kc < 16; ++kc) {
      v16bf bfr = gB(wf + OFF_WD1, kc * 16 + nt, lane);   // W_d1 stays in L2
      acc = wmma_bf16(xf[kc], bfr, acc);
    }
    int n = nt * 16 + ln;
    float bias = b_d1[n];
    float sc = bn_gamma[n] * rsqrtf(bn_var[n] + 1e-3f);
    float sh = bn_beta[n] - bn_mean[n] * sc;
#pragma unroll
    for (int e = 0; e < 8; ++e) {
      float v = (acc[e] + bias) * sc + sh;
      v = (v >= 0.0f) ? v : 0.2f * v;
      dso[(e + 8 * half) * DSO_S + n] = (__bf16)v;  // D-frag: lane owns column n
    }
  }

  // weight DMA must be complete and visible to all waves before LSTM phase
#if HAVE_ASYNC_LDS
  __builtin_amdgcn_s_wait_asynccnt(0);
#endif
  __syncthreads();

  // ---- LSTM states ----
  v8f c1[8], c2[2];
#pragma unroll
  for (int j = 0; j < 8; ++j)
#pragma unroll
    for (int e = 0; e < 8; ++e) c1[j][e] = 0.0f;
#pragma unroll
  for (int j = 0; j < 2; ++j)
#pragma unroll
    for (int e = 0; e < 8; ++e) c2[j][e] = 0.0f;

  for (int t = 0; t < T_STEPS; ++t) {
    // ---- LSTM1 step: z = x_t@Wx1 + h1@Wh1 + b1 ----
    v16bf axt = ldsA(dso, DSO_S, t * 16, false, lane);   // K=16 zero-padded to 32
    v16bf ah[4];
#pragma unroll
    for (int kt = 0; kt < 4; ++kt) ah[kt] = ldsA(h1b, H1_S, kt * 32, true, lane);

#pragma unroll
    for (int j = 0; j < 8; ++j) {        // hidden-unit tile (16 cols)
      v8f acc[4];                        // gates i,f,g,o
#pragma unroll
      for (int g = 0; g < 4; ++g) {
        float bv = b1[g * H1 + j * 16 + ln];
#pragma unroll
        for (int e = 0; e < 8; ++e) acc[g][e] = bv;
      }
#pragma unroll
      for (int g = 0; g < 4; ++g) {
        int nt = g * 8 + j;
        acc[g] = wmma_bf16(axt, ldsB(sm + LW_WX1, nt, lane), acc[g]);
#pragma unroll
        for (int kt = 0; kt < 4; ++kt)
          acc[g] = wmma_bf16(ah[kt], ldsB(sm + LW_WH1, kt * 32 + nt, lane), acc[g]);
      }
#pragma unroll
      for (int e = 0; e < 8; ++e) {
        float iv = sigm(acc[0][e]);
        float fv = sigm(acc[1][e]);
        float gv = fast_tanh(acc[2][e]);
        float ov = sigm(acc[3][e]);
        float c = fv * c1[j][e] + iv * gv;
        c1[j][e] = c;
        float h = ov * fast_tanh(c);
        h1b[(e + 8 * half) * H1_S + j * 16 + ln] = (__bf16)h;  // D->A via LDS
      }
    }

    // ---- LSTM2 step (consumes h1_t immediately; hs1 never hits HBM) ----
    v16bf a2[4];
#pragma unroll
    for (int kt = 0; kt < 4; ++kt) a2[kt] = ldsA(h1b, H1_S, kt * 32, true, lane);
    v16bf ah2 = ldsA(h2b, H2_S, 0, true, lane);

#pragma unroll
    for (int j = 0; j < 2; ++j) {
      v8f acc[4];
#pragma unroll
      for (int g = 0; g < 4; ++g) {
        float bv = b2[g * H2 + j * 16 + ln];
#pragma unroll
        for (int e = 0; e < 8; ++e) acc[g][e] = bv;
      }
#pragma unroll
      for (int g = 0; g < 4; ++g) {
        int nt = g * 2 + j;
#pragma unroll
        for (int kt = 0; kt < 4; ++kt)
          acc[g] = wmma_bf16(a2[kt], ldsB(sm + LW_WX2, kt * 8 + nt, lane), acc[g]);
        acc[g] = wmma_bf16(ah2, ldsB(sm + LW_WH2, nt, lane), acc[g]);
      }
#pragma unroll
      for (int e = 0; e < 8; ++e) {
        float iv = sigm(acc[0][e]);
        float fv = sigm(acc[1][e]);
        float gv = fast_tanh(acc[2][e]);
        float ov = sigm(acc[3][e]);
        float c = fv * c2[j][e] + iv * gv;
        c2[j][e] = c;
        float h = ov * fast_tanh(c);
        h2b[(e + 8 * half) * H2_S + j * 16 + ln] = (__bf16)h;
      }
    }
  }

  // ---- Dense(10) + softmax on h_last (tiny; lanes 0-15 own one row each) ----
  if (lane < 16) {
    int r = lane;
    float hv[32];
#pragma unroll
    for (int k = 0; k < 32; ++k) hv[k] = (float)h2b[r * H2_S + k];
    float lg[10];
    float mx = -1e30f;
#pragma unroll
    for (int c = 0; c < 10; ++c) {
      float s = b_d2[c];
#pragma unroll
      for (int k = 0; k < 32; ++k) s += hv[k] * W_d2[k * 10 + c];
      lg[c] = s;
      mx = fmaxf(mx, s);
    }
    float den = 0.0f;
#pragma unroll
    for (int c = 0; c < 10; ++c) { lg[c] = __expf(lg[c] - mx); den += lg[c]; }
    float inv = 1.0f / den;
    float* orow = out + (size_t)(r0 + r) * 10;
#pragma unroll
    for (int c = 0; c < 10; ++c) orow[c] = lg[c] * inv;
  }
}

// ---------------- host launcher ----------------
extern "C" void kernel_launch(void* const* d_in, const int* in_sizes, int n_in,
                              void* d_out, int out_size, void* d_ws, size_t ws_size,
                              hipStream_t stream) {
  const float* x        = (const float*)d_in[0];
  const float* W_d1     = (const float*)d_in[1];
  const float* b_d1     = (const float*)d_in[2];
  const float* bn_gamma = (const float*)d_in[3];
  const float* bn_beta  = (const float*)d_in[4];
  const float* bn_mean  = (const float*)d_in[5];
  const float* bn_var   = (const float*)d_in[6];
  const float* Wx1      = (const float*)d_in[7];
  const float* Wh1      = (const float*)d_in[8];
  const float* b1       = (const float*)d_in[9];
  const float* Wx2      = (const float*)d_in[10];
  const float* Wh2      = (const float*)d_in[11];
  const float* b2       = (const float*)d_in[12];
  const float* W_d2     = (const float*)d_in[13];
  const float* b_d2     = (const float*)d_in[14];
  float* out = (float*)d_out;
  __bf16* wf = (__bf16*)d_ws;

  (void)in_sizes; (void)n_in; (void)out_size; (void)ws_size;

  (void)hipFuncSetAttribute((const void*)rnn_fused,
                            hipFuncAttributeMaxDynamicSharedMemorySize, SMEM_BYTES);

  prep_weights<<<(TOTAL_FRAG_ELEMS + 255) / 256, 256, 0, stream>>>(W_d1, Wx1, Wh1,
                                                                   Wx2, Wh2, wf);
  rnn_fused<<<B_ROWS / (NWAVES * 16), 256, SMEM_BYTES, stream>>>(
      x, b_d1, bn_gamma, bn_beta, bn_mean, bn_var, b1, b2, W_d2, b_d2, wf, out);
}
